// blockNL_84859963835078
// MI455X (gfx1250) — matrix operand order repaired and checked by reading, not compiled, not run on hardware
//
#include <hip/hip_runtime.h>
#include <hip/hip_bf16.h>

typedef __attribute__((ext_vector_type(16))) _Float16 v16h;
typedef __attribute__((ext_vector_type(8)))  float    v8f;

#define TS   16            // attention pixel tile side
#define HALO 4
#define TH_  (TS + 2*HALO) // 24
#define CHK  16            // channel chunk

// ---------------------------------------------------------------------------
// Generic WMMA GEMM:  out[b, m, p] = sum_k Wmat[m, k] * in[b, k, p] (+ addend)
//   Wmat: [64, K] row-major (f32, staged to LDS as f16)
//   in  : rows stride P, batch stride inBatch (f32)
//   out : row stride outRow, batch stride outBatch (f32)
//   addend (optional): [B, 64, P]
// Block = 256 threads = 8 waves; wave w computes a 64(M) x 16(N) strip.
// ---------------------------------------------------------------------------
__global__ __launch_bounds__(256)
void gemm64_wmma_kernel(const float* __restrict__ Wmat,
                        const float* __restrict__ in,
                        const float* __restrict__ addend,
                        float* __restrict__ outp,
                        int K, int P,
                        long inBatch, long outRow, long outBatch)
{
    extern __shared__ _Float16 sW[];          // 64 * K halves
    const int tid = threadIdx.x;

    // stage weights into LDS as f16
    for (int idx = tid; idx < 64 * K; idx += 256)
        sW[idx] = (_Float16)Wmat[idx];
    __syncthreads();

    const int wave = tid >> 5;
    const int lane = tid & 31;
    const int half = lane >> 4;               // selects K sub-group for A, B
    const int mrow = lane & 15;               // A: M row / B,D: N column
    const int b    = blockIdx.y;
    const int pixBase = blockIdx.x * 128 + wave * 16;

    const float* inB = in + (long)b * inBatch + pixBase + mrow;

    v8f acc[4] = {};                          // 4 M-tiles x 16x16 f32

    for (int kt = 0; kt < K; kt += 32) {
        if (kt + 32 < K)                      // emits global_prefetch_b8
            __builtin_prefetch(inB + (long)(kt + 32 + half * 16) * P, 0, 1);

        // ---- B fragment: 32(K) x 16(N), lanes0-15 K=0..7,16..23, lanes16-31 K=8..15,24..31
        v16h bf;
        #pragma unroll
        for (int v = 0; v < 8; ++v) {
            const int k = kt + 2 * (v & 3) + ((v >> 2) << 4) + half * 8;
            bf[2 * v]     = (_Float16)inB[(long)k * P];
            bf[2 * v + 1] = (_Float16)inB[(long)(k + 1) * P];
        }

        // ---- 4 A fragments (16x32) from LDS, accumulate
        #pragma unroll
        for (int mt = 0; mt < 4; ++mt) {
            const _Float16* wrow = sW + (mt * 16 + mrow) * K + kt + half * 8;
            v16h af;
            #pragma unroll
            for (int v = 0; v < 8; ++v) {
                const int k0 = 2 * (v & 3) + ((v >> 2) << 4); // 0,2,4,6,16,18,20,22
                af[2 * v]     = wrow[k0];
                af[2 * v + 1] = wrow[k0 + 1];
            }
            acc[mt] = __builtin_amdgcn_wmma_f32_16x16x32_f16(
                false, af, false, bf, (short)0, acc[mt], false, false);
        }
    }

    // ---- store D: VGPR r -> M = mt*16 + r + 8*half, N = lane&15
    float* outB = outp + (long)b * outBatch + pixBase + mrow;
    const float* addB = addend ? addend + ((long)b * 64) * P + pixBase + mrow
                               : nullptr;
    #pragma unroll
    for (int mt = 0; mt < 4; ++mt) {
        #pragma unroll
        for (int r = 0; r < 8; ++r) {
            const int row = mt * 16 + r + half * 8;
            float val = acc[mt][r];
            if (addB) val += addB[(long)row * P];
            outB[(long)row * outRow] = val;
        }
    }
}

// ---------------------------------------------------------------------------
// 9x9 windowed attention:
//   att[k] = sum_c theta[c,p] * phi[c, p+off(k)]  (zero-padded)
//   att    = softmax_k(att)
//   o[c,p] = sum_k att[k] * g[c, p+off(k)]
// Block = 16x16 pixel tile (256 threads, one pixel per thread).
// phi/g staged in LDS (f16) as 24x24 halo tiles in 16-channel chunks;
// att scores (81 per pixel) live in LDS (tid-major -> conflict-free).
// ---------------------------------------------------------------------------
__global__ __launch_bounds__(256)
void attn9x9_kernel(const float* __restrict__ theta,
                    const float* __restrict__ phi,
                    const float* __restrict__ g,
                    float* __restrict__ o,
                    int H, int W)
{
    extern __shared__ unsigned char smem[];
    float*    s_att  = (float*)smem;                       // 81 * 256 f32
    _Float16* s_tile = (_Float16*)(smem + 81 * 256 * 4);   // CHK * 24 * 24 f16

    const int P   = H * W;
    const int tid = threadIdx.x;
    const int px  = tid & 15, py = tid >> 4;
    const int b   = blockIdx.z;
    const int w0  = blockIdx.x * TS, h0 = blockIdx.y * TS;
    const int h   = h0 + py, w = w0 + px;
    const int p   = h * W + w;

    const float* thB = theta + (long)b * 64 * P;
    const float* phB = phi   + (long)b * 64 * P;
    const float* gB  = g     + (long)b * 64 * P;
    float*       oB  = o     + (long)b * 64 * P;

    for (int k = 0; k < 81; ++k) s_att[k * 256 + tid] = 0.f;

    // ---------------- attention scores ----------------
    for (int cc = 0; cc < 64; cc += CHK) {
        __syncthreads();
        for (int idx = tid; idx < CHK * TH_ * TH_; idx += 256) {
            const int c  = idx / (TH_ * TH_);
            const int r  = idx % (TH_ * TH_);
            const int gh = h0 - HALO + r / TH_;
            const int gw = w0 - HALO + r % TH_;
            float v = 0.f;
            if (gh >= 0 && gh < H && gw >= 0 && gw < W)
                v = phB[(long)(cc + c) * P + gh * W + gw];
            s_tile[idx] = (_Float16)v;
        }
        __syncthreads();

        float th[CHK];
        #pragma unroll
        for (int c = 0; c < CHK; ++c) th[c] = thB[(long)(cc + c) * P + p];

        for (int k = 0; k < 81; ++k) {
            const int idx = (py + k / 9) * TH_ + (px + k % 9);
            float s = 0.f;
            #pragma unroll
            for (int c = 0; c < CHK; ++c)
                s += th[c] * (float)s_tile[c * TH_ * TH_ + idx];
            s_att[k * 256 + tid] += s;
        }
    }
    __syncthreads();

    // ---------------- softmax over k (per-thread column) ----------------
    float mx = -3.0e38f;
    for (int k = 0; k < 81; ++k) mx = fmaxf(mx, s_att[k * 256 + tid]);
    float sum = 0.f;
    for (int k = 0; k < 81; ++k) {
        const float e = __expf(s_att[k * 256 + tid] - mx);
        s_att[k * 256 + tid] = e;
        sum += e;
    }
    const float inv = 1.f / sum;
    for (int k = 0; k < 81; ++k) s_att[k * 256 + tid] *= inv;

    // ---------------- weighted aggregation of g ----------------
    for (int cc = 0; cc < 64; cc += CHK) {
        __syncthreads();
        for (int idx = tid; idx < CHK * TH_ * TH_; idx += 256) {
            const int c  = idx / (TH_ * TH_);
            const int r  = idx % (TH_ * TH_);
            const int gh = h0 - HALO + r / TH_;
            const int gw = w0 - HALO + r % TH_;
            float v = 0.f;
            if (gh >= 0 && gh < H && gw >= 0 && gw < W)
                v = gB[(long)(cc + c) * P + gh * W + gw];
            s_tile[idx] = (_Float16)v;
        }
        __syncthreads();

        float oreg[CHK] = {};
        for (int k = 0; k < 81; ++k) {
            const float a  = s_att[k * 256 + tid];
            const int  idx = (py + k / 9) * TH_ + (px + k % 9);
            #pragma unroll
            for (int c = 0; c < CHK; ++c)
                oreg[c] += a * (float)s_tile[c * TH_ * TH_ + idx];
        }
        #pragma unroll
        for (int c = 0; c < CHK; ++c)
            oB[(long)(cc + c) * P + p] = oreg[c];
    }
}

// ---------------------------------------------------------------------------
extern "C" void kernel_launch(void* const* d_in, const int* in_sizes, int n_in,
                              void* d_out, int out_size, void* d_ws, size_t ws_size,
                              hipStream_t stream)
{
    const float* x      = (const float*)d_in[0]; // [2,9,64,96,96]
    const float* center = (const float*)d_in[1]; // [2,64,96,96]
    const float* Wt     = (const float*)d_in[2]; // [64,64]
    const float* Wp     = (const float*)d_in[3];
    const float* Wg     = (const float*)d_in[4];
    const float* Ww     = (const float*)d_in[5];
    const float* Wbig   = (const float*)d_in[6]; // [64,576]
    float*       out    = (float*)d_out;         // [2,64,96,96]

    const int B = 2, N = 9, C = 64, H = 96, W = 96;
    const int P  = H * W;        // 9216
    const int CN = C * N;        // 576

    // workspace layout (floats)
    const size_t szCP = (size_t)B * C * P;       // 1,179,648 floats
    float* theta   = (float*)d_ws;
    float* phi     = theta + szCP;
    float* gbuf    = phi   + szCP;
    float* obuf    = gbuf  + szCP;
    float* stacked = obuf  + szCP;               // [B, C*N, P]

    const dim3 gGrid(P / 128, B);                // 72 x 2 blocks, 256 thr
    const dim3 aGrid(W / TS, H / TS, B);         // 6 x 6 x 2 blocks
    const size_t smemW64  = (size_t)64 * C  * sizeof(_Float16); //  8 KB
    const size_t smemWbig = (size_t)64 * CN * sizeof(_Float16); // 72 KB
    const size_t smemAttn = (size_t)81 * 256 * 4 + (size_t)CHK * TH_ * TH_ * 2;

    // theta = Wt . center
    gemm64_wmma_kernel<<<gGrid, 256, smemW64, stream>>>(
        Wt, center, nullptr, theta,
        C, P, (long)C * P, (long)P, (long)64 * P);

    for (int i = 0; i < N; ++i) {
        const float* xi = x + (size_t)i * C * P;   // batch stride N*C*P

        // phi = Wp . x_i ,  g = Wg . x_i
        gemm64_wmma_kernel<<<gGrid, 256, smemW64, stream>>>(
            Wp, xi, nullptr, phi,
            C, P, (long)N * C * P, (long)P, (long)64 * P);
        gemm64_wmma_kernel<<<gGrid, 256, smemW64, stream>>>(
            Wg, xi, nullptr, gbuf,
            C, P, (long)N * C * P, (long)P, (long)64 * P);

        // o = softmax(theta . patches(phi)) . patches(g)
        attn9x9_kernel<<<aGrid, 256, smemAttn, stream>>>(
            theta, phi, gbuf, obuf, H, W);

        // stacked[:, c*N + i, :] = Ww . o + center
        gemm64_wmma_kernel<<<gGrid, 256, smemW64, stream>>>(
            Ww, obuf, center, stacked + (size_t)i * P,
            C, P, (long)C * P, (long)N * P, (long)C * N * P);
    }

    // out = Wbig . stacked   (K = 576)
    gemm64_wmma_kernel<<<gGrid, 256, smemWbig, stream>>>(
        Wbig, stacked, nullptr, out,
        CN, P, (long)CN * P, (long)P, (long)64 * P);
}
